// MultiGaussKernelEE_47794396070557
// MI455X (gfx1250) — matrix operand order repaired and checked by reading, not compiled, not run on hardware
//
#include <hip/hip_runtime.h>
#include <math.h>

typedef __attribute__((ext_vector_type(16))) _Float16 v16h;
typedef __attribute__((ext_vector_type(8)))  float    v8f;
typedef unsigned int u32x4 __attribute__((ext_vector_type(4)));
typedef int          i32x4 __attribute__((ext_vector_type(4)));
typedef int          i32x8 __attribute__((ext_vector_type(8)));

#define N_SAMP 8192
#define KMIX   256
#define DIM    64

// ---------------- workspace layout (bytes) ----------------
// Amat  : f16 [K][64][64]            @ 0         size 2 MiB
// bvec  : f32 [K][64]                @ 2097152   size 64 KiB
// cvec  : f32 [K]                    @ 2162688   size 1 KiB
// wlse  : f32 [1]                    @ 2163712
// rowout: f32 [N]                    @ 2164224   size 32 KiB
#define OFF_B    (2u * 1024u * 1024u)
#define OFF_C    (OFF_B + KMIX * DIM * 4u)
#define OFF_WLSE (OFF_C + KMIX * 4u)
#define OFF_ROW  (OFF_WLSE + 512u)

// -------- P0: log-sum-exp of mixture weights (K == 256) --------
__global__ void k_wlse(const float* __restrict__ weigh, float* __restrict__ wlse) {
    __shared__ float red[256];
    int t = threadIdx.x;
    float v = weigh[t];
    red[t] = v; __syncthreads();
    for (int s = 128; s > 0; s >>= 1) { if (t < s) red[t] = fmaxf(red[t], red[t + s]); __syncthreads(); }
    float mx = red[0]; __syncthreads();
    red[t] = __expf(v - mx); __syncthreads();
    for (int s = 128; s > 0; s >>= 1) { if (t < s) red[t] += red[t + s]; __syncthreads(); }
    if (t == 0) *wlse = mx + logf(red[0]);
}

// -------- P1: build A_k = (I + tril(tri,-1)) * diag(tanh(diag)), b_k, c_k --------
__global__ void k_prep(const float* __restrict__ means, const float* __restrict__ diag,
                       const float* __restrict__ tri,   const float* __restrict__ weigh,
                       const float* __restrict__ wlse,
                       _Float16* __restrict__ Amat, float* __restrict__ bvec,
                       float* __restrict__ cvec) {
    int k = blockIdx.x;
    int i = threadIdx.x;              // 64 threads: one output row each
    __shared__ float lv[DIM];
    __shared__ float mu[DIM];
    lv[i] = tanhf(diag[k * DIM + i]);
    mu[i] = means[k * DIM + i];
    __syncthreads();
    const float* trik = tri + (size_t)k * DIM * DIM;
    _Float16*    Ak   = Amat + (size_t)k * DIM * DIM;
    float b = 0.f;
    for (int j = 0; j < DIM; ++j) {
        float a = ((i == j) ? 1.0f : 0.0f) + ((j < i) ? trik[i * DIM + j] : 0.0f);
        a *= lv[j];
        Ak[i * DIM + j] = (_Float16)a;
        b = fmaf(a, mu[j], b);
    }
    bvec[k * DIM + i] = b;
    if (i == 0) {
        float ld = 0.f;
        for (int j = 0; j < DIM; ++j) ld += logf(fabsf(lv[j]));
        cvec[k] = ld + weigh[k] - wlse[0];
    }
}

// binary select of v[idx], idx in [0,8) — lowers to v_cndmask chain, no scratch
__device__ __forceinline__ float sel8(const float v[8], int idx) {
    float a = (idx & 1) ? v[1] : v[0];
    float b = (idx & 1) ? v[3] : v[2];
    float c = (idx & 1) ? v[5] : v[4];
    float d = (idx & 1) ? v[7] : v[6];
    float e = (idx & 2) ? b : a;
    float f = (idx & 2) ? d : c;
    return (idx & 4) ? f : e;
}

// -------- TDM: DMA one 8 KiB A_k panel (2048 dwords, 1-D tensor) into LDS --------
// D# per CDNA5 ISA; 6-arg builtin form (groups 2/3 + pad group unused).
__device__ __forceinline__ void tdm_load_panel(const _Float16* gsrc, unsigned lds_addr) {
    unsigned long long ga = (unsigned long long)(uintptr_t)gsrc;
    u32x4 g0 = {};
    g0[0] = 1u;                                              // count=1 (valid user D#)
    g0[1] = lds_addr;                                        // lds_addr (bytes)
    g0[2] = (unsigned)ga;                                    // global_addr[31:0]
    g0[3] = (unsigned)((ga >> 32) & 0x1FFFFFFu) | (2u << 30); // global_addr[56:32] | type=2
    i32x8 g1 = {};
    g1[0] = (int)(2u << 16);        // workgroup_mask=0, data_size=2 (4-byte units)
    g1[1] = (int)(2048u << 16);     // tensor_dim0[15:0]=2048 at bit48
    g1[2] = (int)(1u << 16);        // tensor_dim0[31:16]=0, tensor_dim1=1
    g1[3] = (int)(2048u << 16);     // tensor_dim1 hi=0, tile_dim0=2048
    g1[4] = 1;                      // tile_dim1=1, tile_dim2=0
    g1[5] = 2048;                   // tensor_dim0_stride=2048 (low 32)
    g1[6] = 0;
    g1[7] = 0;
    i32x4 z4 = {};
    i32x8 z8 = {};
    __builtin_amdgcn_tensor_load_to_lds(g0, g1, z4, z4, z8, 0);
}

// -------- main: per 16-row wave tile, loop over K mixtures with WMMA + TDM --------
__global__ __launch_bounds__(128) void k_main(const float* __restrict__ x,
                                              const _Float16* __restrict__ Amat,
                                              const float* __restrict__ bvec,
                                              const float* __restrict__ cvec,
                                              float* __restrict__ rowout) {
    __shared__ _Float16 panel[2][DIM * DIM];   // 2 x 8 KiB double buffer

    const int lane    = threadIdx.x & 31;
    const int wave    = threadIdx.x >> 5;
    const int rowbase = blockIdx.x * 64 + wave * 16;
    const int m       = lane & 15;   // row within tile (A frag), col within tile (B/D frag)
    const int g       = lane >> 4;   // lane-half selector
    const bool leader = (threadIdx.x < 32);    // wave 0 drives the TDM

    const unsigned lds0 = (unsigned)(uintptr_t)&panel[0][0];
    const unsigned lds1 = (unsigned)(uintptr_t)&panel[1][0];

    // ---- build the two A fragments (X tile rows, fixed across the whole k loop) ----
    // 16-bit A 16x32 layout (contiguous, matches B): half h -> K = g*16+h.
    v16h afrag[2];
    const float* xrow = x + (size_t)(rowbase + m) * DIM;
    #pragma unroll
    for (int s = 0; s < 2; ++s) {
        const int b1 = s * 32 + g * 16;
        const int b2 = s * 32 + g * 16 + 8;
        #pragma unroll
        for (int h = 0; h < 8; ++h) afrag[s][h]     = (_Float16)xrow[b1 + h];
        #pragma unroll
        for (int h = 0; h < 8; ++h) afrag[s][8 + h] = (_Float16)xrow[b2 + h];
    }

    // prologue: DMA panel 0 into buffer 0
    if (leader) tdm_load_panel(Amat, lds0);

    float mx = -__builtin_inff();
    float sm = 0.0f;

    for (int k = 0; k < KMIX; ++k) {
        if (leader) __builtin_amdgcn_s_wait_tensorcnt(0);
        __syncthreads();   // panel k resident; all waves done reading the other buffer

        if (leader && (k + 1 < KMIX))
            tdm_load_panel(Amat + (size_t)(k + 1) * DIM * DIM, (k & 1) ? lds0 : lds1);

        const _Float16* Ak = &panel[k & 1][0];

        float sq[8];
        #pragma unroll
        for (int r = 0; r < 8; ++r) sq[r] = 0.f;

        #pragma unroll
        for (int nt = 0; nt < 4; ++nt) {
            // B fragment (32x16 f16): VGPR v, lanes<16 hold K=2v,2v+1 ; lanes>=16 hold K=16+2v,+17;
            // lane%16 = column n. B[j][n] = A_k[nt*16+n][j] -> 16 contiguous halfs of row (nt*16+m)
            // starting at column kstep*32 + g*16 (32-byte aligned -> ds_load_b128 x2).
            const v16h b0 = *(const v16h*)(Ak + (nt * 16 + m) * DIM + g * 16);
            const v16h b1 = *(const v16h*)(Ak + (nt * 16 + m) * DIM + 32 + g * 16);
            v8f acc = {};
            acc = __builtin_amdgcn_wmma_f32_16x16x32_f16(false, afrag[0], false, b0,
                                                         (short)0, acc, false, false);
            acc = __builtin_amdgcn_wmma_f32_16x16x32_f16(false, afrag[1], false, b1,
                                                         (short)0, acc, false, false);
            const float bv = bvec[k * DIM + nt * 16 + m];
            #pragma unroll
            for (int r = 0; r < 8; ++r) { float z = acc[r] - bv; sq[r] = fmaf(z, z, sq[r]); }
        }

        // reduce squares over the 16 lanes of each half (columns of the D tiles)
        #pragma unroll
        for (int off = 1; off < 16; off <<= 1) {
            #pragma unroll
            for (int r = 0; r < 8; ++r) sq[r] += __shfl_xor(sq[r], off, 32);
        }
        // lower half now has rows 0..7, upper half rows 8..15; swap copies across halves
        float tq[8];
        #pragma unroll
        for (int r = 0; r < 8; ++r) tq[r] = __shfl_xor(sq[r], 16, 32);

        const bool useS = ((m >> 3) == g);
        const int  r    = m & 7;
        const float q   = useS ? sel8(sq, r) : sel8(tq, r);

        // online log-sum-exp over k
        const float val = cvec[k] - 0.5f * q;
        const float nm  = fmaxf(mx, val);
        sm = sm * __expf(mx - nm) + __expf(val - nm);
        mx = nm;
    }

    if (lane < 16) {
        const float logC = -32.0f * 1.8378770664093453f;   // -(d/2)*ln(2*pi), d=64
        rowout[rowbase + m] = -(logC + mx + logf(sm));
    }
}

// -------- deterministic final mean over N --------
__global__ void k_reduce(const float* __restrict__ rowout, float* __restrict__ out) {
    __shared__ float red[256];
    int t = threadIdx.x;
    float s = 0.f;
    for (int i = t; i < N_SAMP; i += 256) s += rowout[i];
    red[t] = s; __syncthreads();
    for (int st = 128; st > 0; st >>= 1) { if (t < st) red[t] += red[t + st]; __syncthreads(); }
    if (t == 0) out[0] = red[0] * (1.0f / N_SAMP);
}

extern "C" void kernel_launch(void* const* d_in, const int* in_sizes, int n_in,
                              void* d_out, int out_size, void* d_ws, size_t ws_size,
                              hipStream_t stream) {
    const float* x     = (const float*)d_in[0];   // [8192,64]
    const float* means = (const float*)d_in[1];   // [256,64]
    const float* diag  = (const float*)d_in[2];   // [1,256,64]
    const float* tri   = (const float*)d_in[3];   // [1,256,64,64]
    const float* weigh = (const float*)d_in[4];   // [1,256]
    float*       out   = (float*)d_out;

    char* ws = (char*)d_ws;
    _Float16* Amat   = (_Float16*)(ws);
    float*    bvec   = (float*)(ws + OFF_B);
    float*    cvec   = (float*)(ws + OFF_C);
    float*    wlse   = (float*)(ws + OFF_WLSE);
    float*    rowout = (float*)(ws + OFF_ROW);

    k_wlse  <<<1,    256, 0, stream>>>(weigh, wlse);
    k_prep  <<<KMIX, DIM, 0, stream>>>(means, diag, tri, weigh, wlse, Amat, bvec, cvec);
    k_main  <<<N_SAMP / 64, 128, 0, stream>>>(x, Amat, bvec, cvec, rowout);
    k_reduce<<<1,    256, 0, stream>>>(rowout, out);
}
